// LatentSelfAttention_29738353557790
// MI455X (gfx1250) — matrix operand order repaired and checked by reading, not compiled, not run on hardware
//
#include <hip/hip_runtime.h>
#include <hip/hip_bf16.h>

// ---------------------------------------------------------------------------
// Problem constants (fixed by the reference): B=4, s=2048, P=2048, E=L=1024,
// S = P + s = 4096.
// ---------------------------------------------------------------------------
static constexpr int BB   = 4;
static constexpr int SLEN = 2048;
static constexpr int PLEN = 2048;
static constexpr int EE   = 1024;
static constexpr int LL   = 1024;
static constexpr int SS   = PLEN + SLEN;   // 4096

typedef __attribute__((ext_vector_type(16))) __bf16         bf16x16;
typedef __attribute__((ext_vector_type(8)))  __bf16         bf16x8;
typedef __attribute__((ext_vector_type(8)))  float          f32x8;
typedef __attribute__((address_space(3)))    unsigned short lds_ushort;

union V16frag { bf16x16 v; bf16x8 h[2]; };

__device__ __forceinline__ unsigned short f32_to_bf16(float f) {
    union { float f; unsigned int u; } v; v.f = f;
    unsigned int lsb = (v.u >> 16) & 1u;
    v.u += 0x7fffu + lsb;                 // round-to-nearest-even
    return (unsigned short)(v.u >> 16);
}

// ---------------------------------------------------------------------------
// CDNA5 async global->LDS copy (ASYNCcnt-tracked), per-lane 16B transfer.
// VDST VGPR = LDS byte offset (AS3 pointer value), VADDR = 64-bit global addr.
// ---------------------------------------------------------------------------
__device__ __forceinline__ void async_load_b128(lds_ushort* lp, const void* g) {
    asm volatile("global_load_async_to_lds_b128 %0, %1, off"
                 :: "v"(lp), "v"(g) : "memory");
}
template <int N>
__device__ __forceinline__ void wait_async() {
    asm volatile("s_wait_asynccnt %0" :: "n"(N) : "memory");
}

// ---------------------------------------------------------------------------
// Generic bf16 WMMA GEMM:  C = alpha * A * op(B) (+ C if beta)
//   A: (M,K) row-major bf16
//   BT=true : B is (N,K) row-major (NT gemm);  BT=false: B is (K,N) row-major
//   OUT_BF16: store bf16, else f32 (beta accumulation only on f32 path)
// Block tile 128x128, K-tile 32, double-buffered LDS staged with
// global_load_async_to_lds_b128. 8 wave32 waves, each wave owns a 32x64 tile
// (2x4 WMMA tiles of v_wmma_f32_16x16x32_bf16 -> 8 WMMA per K-step).
// M%128==0, N%128==0, K%32==0 (guaranteed by problem sizes).
// ---------------------------------------------------------------------------
template <bool BT, bool OUT_BF16>
__global__ __launch_bounds__(256) void gemm_wmma_kernel(
    const unsigned short* __restrict__ A, const unsigned short* __restrict__ B,
    void* __restrict__ Cv, int M, int N, int K, float alpha, int beta,
    long strideA, long strideB, long strideC) {
    __shared__ __align__(16) unsigned short sA[2][128 * 40];
    __shared__ __align__(16) unsigned short sB[2][128 * 40];

    const int tid  = threadIdx.x;
    const int lane = tid & 31;
    const int wave = tid >> 5;
    const int wm   = wave >> 1;      // 0..3 -> 32-row slice
    const int wn   = wave & 1;       // 0..1 -> 64-col slice
    const long z   = blockIdx.z;

    A += z * strideA;
    B += z * strideB;

    const int blockM = blockIdx.y * 128;
    const int blockN = blockIdx.x * 128;

    f32x8 acc[2][4] = {};

    // 256 threads stage a 128x32 tile: 2 threads/row, 16 elems (2x b128) each.
    const int arow = tid >> 1, acb = (tid & 1) * 16;

    auto stage = [&](int buf, int k0) {
        const unsigned short* ga = A + (long)(blockM + arow) * K + k0 + acb;
        async_load_b128((lds_ushort*)&sA[buf][arow * 40 + acb], ga);
        async_load_b128((lds_ushort*)&sA[buf][arow * 40 + acb + 8], ga + 8);
        if (BT) {
            const unsigned short* gb = B + (long)(blockN + arow) * K + k0 + acb;
            async_load_b128((lds_ushort*)&sB[buf][arow * 40 + acb], gb);
            async_load_b128((lds_ushort*)&sB[buf][arow * 40 + acb + 8], gb + 8);
        } else {
            // gather B[k][n] column-strided -> LDS row n (transposed store)
            const unsigned short* gb = B + (long)(k0 + acb) * N + blockN + arow;
            __builtin_prefetch(gb + 32L * N, 0, 1);
#pragma unroll
            for (int i = 0; i < 16; ++i)
                sB[buf][arow * 40 + acb + i] = gb[(long)i * N];
        }
    };

    int buf = 0;
    stage(0, 0);
    for (int k0 = 0; k0 < K; k0 += 32) {
        const bool more = (k0 + 32) < K;
        if (more) stage(buf ^ 1, k0 + 32);
        // wait only for the *current* buffer's async loads (per-wave count:
        // 4 async/stage for NT, 2 for NN), keeping the next stage in flight.
        if (more) { if (BT) wait_async<4>(); else wait_async<2>(); }
        else      { wait_async<0>(); }
        __syncthreads();

        // ---- build WMMA fragments per documented VGPR layouts ----
        const int half = lane >> 4;     // 0: lanes 0-15, 1: lanes 16-31
        const int ar   = lane & 15;
        V16frag a0, a1, b[4];
        {
            const bf16x8* p = (const bf16x8*)(sA[buf] + (wm * 32 + ar) * 40);
            a0.h[0] = p[half];  a0.h[1] = p[2 + half];   // K {0-7,16-23}/{8-15,24-31}
            const bf16x8* q = (const bf16x8*)(sA[buf] + (wm * 32 + 16 + ar) * 40);
            a1.h[0] = q[half];  a1.h[1] = q[2 + half];
#pragma unroll
            for (int j = 0; j < 4; ++j) {
                const bf16x8* r = (const bf16x8*)(sB[buf] + (wn * 64 + j * 16 + ar) * 40);
                b[j].h[0] = r[half];  b[j].h[1] = r[2 + half];  // K {0-7,16-23}/{8-15,24-31}
            }
        }
#pragma unroll
        for (int j = 0; j < 4; ++j) {
            acc[0][j] = __builtin_amdgcn_wmma_f32_16x16x32_bf16(false, a0.v, false, b[j].v, (short)0, acc[0][j], false, false);
            acc[1][j] = __builtin_amdgcn_wmma_f32_16x16x32_bf16(false, a1.v, false, b[j].v, (short)0, acc[1][j], false, false);
        }
        __syncthreads();
        buf ^= 1;
    }

    // ---- epilogue (C/D layout: VGPR p -> M = p + 8*(lane>=16), N = lane&15) ----
    const int lrow = (lane >> 4) * 8;
    const int lcol = lane & 15;
#pragma unroll
    for (int i = 0; i < 2; ++i)
#pragma unroll
        for (int j = 0; j < 4; ++j)
#pragma unroll
            for (int p = 0; p < 8; ++p) {
                int  m   = blockM + wm * 32 + i * 16 + lrow + p;
                int  n   = blockN + wn * 64 + j * 16 + lcol;
                long idx = (long)m * N + n;
                float v  = alpha * acc[i][j][p];
                if (OUT_BF16) {
                    unsigned short* C = (unsigned short*)Cv + z * strideC;
                    C[idx] = f32_to_bf16(v);
                } else {
                    float* C = (float*)Cv + z * strideC;
                    if (beta) v += C[idx];
                    C[idx] = v;
                }
            }
}

// ---------------------------------------------------------------------------
// Elementwise kernels
// ---------------------------------------------------------------------------
__global__ void cvt_f32_bf16_kernel(const float* __restrict__ in,
                                    unsigned short* __restrict__ out, long n) {
    long i = (long)blockIdx.x * blockDim.x + threadIdx.x;
    if (i < n) out[i] = f32_to_bf16(in[i]);
}

// Copy (B,P,L) f32 cache into the (B,S,L) outputs (rows 0..P-1), f32 + bf16.
__global__ void cache_copy_kernel(const float* __restrict__ cache,
                                  float* __restrict__ out_f32,
                                  unsigned short* __restrict__ out_b16) {
    long i = (long)blockIdx.x * blockDim.x + threadIdx.x;
    const long n = (long)BB * PLEN * LL;
    if (i >= n) return;
    int  l = (int)(i % LL);
    long r = i / LL;
    int  t = (int)(r % PLEN);
    int  b = (int)(r / PLEN);
    long o = ((long)b * SS + t) * LL + l;
    float v = cache[i];
    out_f32[o] = v;
    out_b16[o] = f32_to_bf16(v);
}

// LayerNorm over L; writes new rows P..S-1 of c_kv (f32 out + bf16 copy) and
// contiguous n_c_kv bf16 for the q_r GEMM.
__global__ __launch_bounds__(256) void ln_fused_kernel(
    const float* __restrict__ c, const float* __restrict__ g,
    const float* __restrict__ b, float* __restrict__ ckv_f32,
    unsigned short* __restrict__ ckv_b16, unsigned short* __restrict__ ncv_b16) {
    int row = blockIdx.x;               // 0 .. B*s-1
    int bt  = row / SLEN, t = row % SLEN;
    const float* x = c + (long)row * LL;
    __shared__ float red[256];
    int tid = threadIdx.x;

    float s = 0.f;
    for (int l = tid; l < LL; l += 256) s += x[l];
    red[tid] = s; __syncthreads();
    for (int o = 128; o > 0; o >>= 1) { if (tid < o) red[tid] += red[tid + o]; __syncthreads(); }
    float mu = red[0] / LL; __syncthreads();

    float vs = 0.f;
    for (int l = tid; l < LL; l += 256) { float d = x[l] - mu; vs += d * d; }
    red[tid] = vs; __syncthreads();
    for (int o = 128; o > 0; o >>= 1) { if (tid < o) red[tid] += red[tid + o]; __syncthreads(); }
    float inv = rsqrtf(red[0] / LL + 1e-5f);

    long ob = ((long)bt * SS + PLEN + t) * (long)LL;
    for (int l = tid; l < LL; l += 256) {
        float y = (x[l] - mu) * inv * g[l] + b[l];
        ckv_f32[ob + l] = y;
        unsigned short h = f32_to_bf16(y);
        ckv_b16[ob + l] = h;
        ncv_b16[(long)row * LL + l] = h;
    }
}

// RoPE for k_r: in = (B,s,L) f32; write rows P..S-1 of k_r (f32 + bf16).
__global__ void rope_k_kernel(const float* __restrict__ in,
                              float* __restrict__ out_f32,
                              unsigned short* __restrict__ out_b16) {
    const int hd = LL / 2;
    long i = (long)blockIdx.x * blockDim.x + threadIdx.x;
    const long n = (long)BB * SLEN * hd;
    if (i >= n) return;
    int  j = (int)(i % hd);
    long r = i / hd;
    int  t = (int)(r % SLEN);
    int  b = (int)(r / SLEN);
    float ang = t * __powf(10000.f, (-2.f * j) / (float)LL);
    float sn = __sinf(ang), cs = __cosf(ang);
    const float* x = in + r * LL;
    float x1 = x[j], x2 = x[j + hd];
    float o1 = x1 * cs - x2 * sn;
    float o2 = x2 * cs + x1 * sn;
    long ob = ((long)b * SS + PLEN + t) * (long)LL;
    out_f32[ob + j] = o1;      out_f32[ob + j + hd] = o2;
    out_b16[ob + j] = f32_to_bf16(o1);
    out_b16[ob + j + hd] = f32_to_bf16(o2);
}

// RoPE for q_r: in = (B,s,E) f32 pre-activation; add bias bqr, rotate, bf16 out.
__global__ void rope_q_kernel(const float* __restrict__ in,
                              const float* __restrict__ bias,
                              unsigned short* __restrict__ out_b16) {
    const int hd = EE / 2;
    long i = (long)blockIdx.x * blockDim.x + threadIdx.x;
    const long n = (long)BB * SLEN * hd;
    if (i >= n) return;
    int  j = (int)(i % hd);
    long r = i / hd;
    int  t = (int)((r % SLEN));
    float ang = t * __powf(10000.f, (-2.f * j) / (float)EE);
    float sn = __sinf(ang), cs = __cosf(ang);
    const float* x = in + r * EE;
    float x1 = x[j] + bias[j];
    float x2 = x[j + hd] + bias[j + hd];
    out_b16[r * EE + j]      = f32_to_bf16(x1 * cs - x2 * sn);
    out_b16[r * EE + j + hd] = f32_to_bf16(x2 * cs + x1 * sn);
}

// Causal-masked softmax over rows of (B,s,S) f32 scores -> bf16 probs.
// Row m (within s) may attend to t <= P + m.
__global__ __launch_bounds__(256) void softmax_kernel(
    const float* __restrict__ scores, unsigned short* __restrict__ probs) {
    int row = blockIdx.x;               // 0 .. B*s-1
    int m = row % SLEN;
    int allowed = PLEN + m + 1;
    const float* x = scores + (long)row * SS;
    __shared__ float red[256];
    int tid = threadIdx.x;

    float mx = -3.4e38f;
    for (int t = tid; t < allowed; t += 256) mx = fmaxf(mx, x[t]);
    red[tid] = mx; __syncthreads();
    for (int o = 128; o > 0; o >>= 1) { if (tid < o) red[tid] = fmaxf(red[tid], red[tid + o]); __syncthreads(); }
    mx = red[0]; __syncthreads();

    float sum = 0.f;
    for (int t = tid; t < allowed; t += 256) sum += __expf(x[t] - mx);
    red[tid] = sum; __syncthreads();
    for (int o = 128; o > 0; o >>= 1) { if (tid < o) red[tid] += red[tid + o]; __syncthreads(); }
    float inv = 1.f / red[0];

    unsigned short* o = probs + (long)row * SS;
    for (int t = tid; t < SS; t += 256) {
        float p = (t < allowed) ? __expf(x[t] - mx) * inv : 0.f;
        o[t] = f32_to_bf16(p);
    }
}

// ---------------------------------------------------------------------------
// Host orchestration
// ---------------------------------------------------------------------------
extern "C" void kernel_launch(void* const* d_in, const int* in_sizes, int n_in,
                              void* d_out, int out_size, void* d_ws, size_t ws_size,
                              hipStream_t stream) {
    (void)in_sizes; (void)n_in; (void)out_size; (void)ws_size;
    const float* x    = (const float*)d_in[0];
    const float* kvc  = (const float*)d_in[1];
    const float* krc  = (const float*)d_in[2];
    const float* Wq   = (const float*)d_in[3];
    const float* Wdkv = (const float*)d_in[4];
    const float* Wuk  = (const float*)d_in[5];
    const float* Wuv  = (const float*)d_in[6];
    const float* Wo   = (const float*)d_in[7];
    const float* Wkr  = (const float*)d_in[8];
    const float* Wqr  = (const float*)d_in[9];
    const float* bqr  = (const float*)d_in[10];
    const float* ln_g = (const float*)d_in[11];
    const float* ln_b = (const float*)d_in[12];

    const long MX   = (long)BB * SLEN;        // 8192 query rows
    const long MT   = (long)BB * SS;          // 16384 key rows
    float* out_o   = (float*)d_out;                      // (B,s,E)
    float* out_ckv = out_o + MX * EE;                    // (B,S,L)
    float* out_kr  = out_ckv + MT * LL;                  // (B,S,L)

    // ---- workspace layout (bf16 stored as ushort) ----
    char* ws = (char*)d_ws;
    size_t off = 0;
    auto take = [&](size_t bytes) -> char* {
        char* p = ws + off;
        off += (bytes + 255) & ~(size_t)255;
        return p;
    };
    unsigned short* x_b    = (unsigned short*)take(MX * EE * 2);
    unsigned short* wq_b   = (unsigned short*)take((size_t)EE * EE * 2);
    unsigned short* wuk_b  = (unsigned short*)take((size_t)LL * EE * 2);
    unsigned short* wdkv_b = (unsigned short*)take((size_t)LL * EE * 2);
    unsigned short* wkr_b  = (unsigned short*)take((size_t)LL * EE * 2);
    unsigned short* wqr_b  = (unsigned short*)take((size_t)EE * LL * 2);
    unsigned short* wuv_b  = (unsigned short*)take((size_t)LL * EE * 2);
    unsigned short* wo_b   = (unsigned short*)take((size_t)EE * EE * 2);
    unsigned short* abs_b  = (unsigned short*)take((size_t)EE * LL * 2);
    unsigned short* ncv_b  = (unsigned short*)take(MX * LL * 2);
    unsigned short* qr_b   = (unsigned short*)take(MX * EE * 2);
    unsigned short* xa_b   = (unsigned short*)take(MX * LL * 2);
    unsigned short* ckv_b  = (unsigned short*)take(MT * LL * 2);
    unsigned short* kr_b   = (unsigned short*)take(MT * LL * 2);
    unsigned short* val_b  = (unsigned short*)take(MT * EE * 2);
    unsigned short* opre_b = (unsigned short*)take(MX * EE * 2);
    float*          f32scr = (float*)take(MX * LL * 4);          // reused 3x
    float*          scores = (float*)take(MX * (size_t)SS * 4);
    unsigned short* probs  = (unsigned short*)take(MX * (size_t)SS * 2);

    const float scale = 0.0220970869120796f;  // 1/sqrt(2*L)

    auto cvt = [&](const float* src, unsigned short* dst, long n) {
        cvt_f32_bf16_kernel<<<dim3((unsigned)((n + 255) / 256)), 256, 0, stream>>>(src, dst, n);
    };

    // 1. bf16 conversions
    cvt(x, x_b, MX * EE);
    cvt(Wq, wq_b, (long)EE * EE);
    cvt(Wuk, wuk_b, (long)LL * EE);
    cvt(Wdkv, wdkv_b, (long)LL * EE);
    cvt(Wkr, wkr_b, (long)LL * EE);
    cvt(Wqr, wqr_b, (long)EE * LL);
    cvt(Wuv, wuv_b, (long)LL * EE);
    cvt(Wo, wo_b, (long)EE * EE);

    // 2. cache concat (rows 0..P-1 of c_kv / k_r outputs + bf16 copies)
    {
        long n = (long)BB * PLEN * LL;
        dim3 g((unsigned)((n + 255) / 256));
        cache_copy_kernel<<<g, 256, 0, stream>>>(kvc, out_ckv, ckv_b);
        cache_copy_kernel<<<g, 256, 0, stream>>>(krc, out_kr, kr_b);
    }

    // 3. absorbed = Wq @ Wuk^T   (E x L), bf16 out
    gemm_wmma_kernel<true, true><<<dim3(LL / 128, EE / 128, 1), 256, 0, stream>>>(
        wq_b, wuk_b, abs_b, EE, LL, EE, 1.f, 0, 0, 0, 0);

    // 4. c = x @ Wdkv^T -> f32 scratch; 5. LayerNorm
    gemm_wmma_kernel<true, false><<<dim3(LL / 128, (unsigned)(MX / 128), 1), 256, 0, stream>>>(
        x_b, wdkv_b, f32scr, (int)MX, LL, EE, 1.f, 0, 0, 0, 0);
    ln_fused_kernel<<<dim3((unsigned)MX), 256, 0, stream>>>(
        f32scr, ln_g, ln_b, out_ckv, ckv_b, ncv_b);

    // 6. k_r pre = x @ Wkr^T -> f32 scratch; 7. RoPE -> rows P.. of k_r
    gemm_wmma_kernel<true, false><<<dim3(LL / 128, (unsigned)(MX / 128), 1), 256, 0, stream>>>(
        x_b, wkr_b, f32scr, (int)MX, LL, EE, 1.f, 0, 0, 0, 0);
    {
        long n = MX * (LL / 2);
        rope_k_kernel<<<dim3((unsigned)((n + 255) / 256)), 256, 0, stream>>>(f32scr, out_kr, kr_b);
    }

    // 8. q pre = n_c_kv @ Wqr^T -> f32 scratch; 9. +bias, RoPE -> q_r bf16
    gemm_wmma_kernel<true, false><<<dim3(EE / 128, (unsigned)(MX / 128), 1), 256, 0, stream>>>(
        ncv_b, wqr_b, f32scr, (int)MX, EE, LL, 1.f, 0, 0, 0, 0);
    {
        long n = MX * (EE / 2);
        rope_q_kernel<<<dim3((unsigned)((n + 255) / 256)), 256, 0, stream>>>(f32scr, bqr, qr_b);
    }

    // 10. xa = x @ absorbed   (NN, bf16 out)
    gemm_wmma_kernel<false, true><<<dim3(LL / 128, (unsigned)(MX / 128), 1), 256, 0, stream>>>(
        x_b, abs_b, xa_b, (int)MX, LL, EE, 1.f, 0, 0, 0, 0);

    // 11. scores = scale*(xa @ c_kv^T) + scale*(q_r @ k_r^T)  (batched NT, f32)
    gemm_wmma_kernel<true, false><<<dim3(SS / 128, SLEN / 128, BB), 256, 0, stream>>>(
        xa_b, ckv_b, scores, SLEN, SS, LL, scale, 0,
        (long)SLEN * LL, (long)SS * LL, (long)SLEN * SS);
    gemm_wmma_kernel<true, false><<<dim3(SS / 128, SLEN / 128, BB), 256, 0, stream>>>(
        qr_b, kr_b, scores, SLEN, SS, EE, scale, 1,
        (long)SLEN * EE, (long)SS * LL, (long)SLEN * SS);

    // 12. masked softmax -> bf16 probs
    softmax_kernel<<<dim3((unsigned)MX), 256, 0, stream>>>(scores, probs);

    // 13. value = c_kv @ Wuv  (NN, bf16 out, all batches flattened)
    gemm_wmma_kernel<false, true><<<dim3(EE / 128, (unsigned)(MT / 128), 1), 256, 0, stream>>>(
        ckv_b, wuv_b, val_b, (int)MT, EE, LL, 1.f, 0, 0, 0, 0);

    // 14. out_pre = probs @ value  (batched NN, bf16 out)
    gemm_wmma_kernel<false, true><<<dim3(EE / 128, SLEN / 128, BB), 256, 0, stream>>>(
        probs, val_b, opre_b, SLEN, EE, SS, 1.f, 0,
        (long)SLEN * SS, (long)SS * EE, (long)SLEN * EE);

    // 15. out = out_pre @ Wo^T  (NT, f32 -> d_out)
    gemm_wmma_kernel<true, false><<<dim3(EE / 128, (unsigned)(MX / 128), 1), 256, 0, stream>>>(
        opre_b, wo_b, out_o, (int)MX, EE, EE, 1.f, 0, 0, 0, 0);
}
